// KANLayer_88802743812698
// MI455X (gfx1250) — matrix operand order repaired and checked by reading, not compiled, not run on hardware
//
#include <hip/hip_runtime.h>
#include <hip/hip_bf16.h>
#include <math.h>

// ---------------------------------------------------------------------------
// KAN layer:  out[b,o] = sum_i rw[o,i]*silu(x[b,i]) + uw[o,i]*sum_j coef[o,i,j]*Bj(x[b,i])
// Reformulated as one GEMM: M=128 (batch), N=256 (out), K=256*12=3072
//   A[b, i*12+j] = j<11 ? basis_j(x[b,i]) : silu(x[b,i])
//   W[i*12+j, o] = j<11 ? coef[o,i,j]*uw[o,i] : rw[o,i]
// bf16 hi/lo split (3 WMMA passes) for fp32-grade accuracy on the bf16 pipe.
// ---------------------------------------------------------------------------

#define B_ROWS   128
#define IN_D     256
#define OUT_D    256
#define NCOEF    11
#define FEATS    12
#define KTOT     (IN_D * FEATS)        // 3072
#define KC_TOT   (KTOT / 32)           // 96 k-chunks of 32
#define TM_TILES (B_ROWS / 16)         // 8
#define TN_TILES (OUT_D / 16)          // 16
#define NTILES   (TM_TILES * TN_TILES) // 128
#define KSPLIT   4
#define KC_PER   (KC_TOT / KSPLIT)     // 24

typedef __attribute__((ext_vector_type(16))) __bf16 v16bf;
typedef __attribute__((ext_vector_type(8)))  float  v8f;

// workspace byte offsets (all 32B aligned)
#define OFF_WHI  0u
#define OFF_WLO  1572864u   // 16*96*512 ushorts = 1.5 MB each
#define OFF_AHI  3145728u
#define OFF_ALO  3932160u   // 8*96*512 ushorts = 0.75 MB each
#define OFF_PART 4718592u   // 4*128*256 floats = 0.5 MB

__device__ __forceinline__ unsigned short f32_to_bf16_rne(float f) {
    unsigned int u = __float_as_uint(f);
    unsigned int r = u + 0x7FFFu + ((u >> 16) & 1u);   // round-to-nearest-even
    return (unsigned short)(r >> 16);
}
__device__ __forceinline__ float bf16_bits_to_f32(unsigned short h) {
    return __uint_as_float(((unsigned int)h) << 16);
}

// ---------------------------------------------------------------------------
// Weights -> pre-swizzled B-matrix fragments (hi/lo bf16).
// B 32x16 fragment in contiguous k order (as A): lane = (k%32 >= 16)*16 + (n%16); element e = k%16.
// Fragment (tn, kc) stored contiguously: 32 lanes x 16 bf16 = 1 KB.
// ---------------------------------------------------------------------------
__global__ void kan_prep_w(const float* __restrict__ coef,
                           const float* __restrict__ rw,
                           const float* __restrict__ uw,
                           unsigned short* __restrict__ whi,
                           unsigned short* __restrict__ wlo) {
    int idx = blockIdx.x * blockDim.x + threadIdx.x;   // OUT_D*IN_D = 65536
    int n = idx >> 8;
    int i = idx & 255;
    float f[FEATS];
    const float* cp = coef + ((size_t)n * IN_D + i) * NCOEF;
    float u = uw[n * IN_D + i];
#pragma unroll
    for (int j = 0; j < NCOEF; ++j) f[j] = cp[j] * u;
    f[11] = rw[n * IN_D + i];

    int tn = n >> 4;
    int nl = n & 15;
#pragma unroll
    for (int j = 0; j < FEATS; ++j) {
        int k    = i * FEATS + j;
        int kc   = k >> 5;
        int kl   = k & 31;
        int lane = ((kl >> 4) << 4) + nl;
        int e    = kl & 15;
        int off  = ((tn * KC_TOT + kc) * 32 + lane) * 16 + e;
        unsigned short hi = f32_to_bf16_rne(f[j]);
        float lof = f[j] - bf16_bits_to_f32(hi);
        whi[off] = hi;
        wlo[off] = f32_to_bf16_rne(lof);
    }
}

// ---------------------------------------------------------------------------
// Activations: cubic B-spline bases + silu -> A-matrix fragments (hi/lo bf16).
// A 16x32 fragment in contiguous k order (as B): lane = (k%32 >= 16)*16 + (m%16),
// element e = k%16.
// ---------------------------------------------------------------------------
__global__ void kan_prep_a(const float* __restrict__ x,
                           const float* __restrict__ grid,
                           unsigned short* __restrict__ ahi,
                           unsigned short* __restrict__ alo) {
    int idx = blockIdx.x * blockDim.x + threadIdx.x;   // B_ROWS*IN_D = 32768
    int b = idx >> 8;
    int i = idx & 255;
    float xv = x[b * IN_D + i];

    float g[15];
#pragma unroll
    for (int t = 0; t < 15; ++t) g[t] = grid[t];  // grid[0][0][:] == shared 1-D knots

    float bas[14];
#pragma unroll
    for (int s = 0; s < 14; ++s)
        bas[s] = (xv >= g[s] && xv < g[s + 1]) ? 1.0f : 0.0f;
#pragma unroll
    for (int p = 1; p <= 3; ++p) {
#pragma unroll
        for (int s = 0; s < 14 - p; ++s) {
            float left  = (xv - g[s]) / (g[s + p] - g[s]) * bas[s];
            float right = (g[s + p + 1] - xv) / (g[s + p + 1] - g[s + 1]) * bas[s + 1];
            bas[s] = left + right;
        }
    }

    float f[FEATS];
#pragma unroll
    for (int j = 0; j < NCOEF; ++j) f[j] = bas[j];
    f[11] = xv / (1.0f + __expf(-xv));   // silu

    int tm = b >> 4;
    int ml = b & 15;
#pragma unroll
    for (int j = 0; j < FEATS; ++j) {
        int k  = i * FEATS + j;
        int kc = k >> 5;
        int kl = k & 31;
        int eh = (kl >> 4) & 1;
        int hl = (kl >> 3) & 1;
        int el = kl & 7;
        int e    = kl & 15;
        int lane = ((kl >> 4) << 4) + ml;
        int off  = ((tm * KC_TOT + kc) * 32 + lane) * 16 + e;
        unsigned short hi = f32_to_bf16_rne(f[j]);
        float lof = f[j] - bf16_bits_to_f32(hi);
        ahi[off] = hi;
        alo[off] = f32_to_bf16_rne(lof);
    }
}

// ---------------------------------------------------------------------------
// WMMA GEMM, split-K=4: one wave owns one 16x16 tile x 24 k-chunks.
// 3 WMMAs/chunk: Ah*Wh + Ah*Wl + Al*Wh  (error ~2^-18 per product).
// ---------------------------------------------------------------------------
__global__ void kan_wmma_gemm(const unsigned short* __restrict__ whi,
                              const unsigned short* __restrict__ wlo,
                              const unsigned short* __restrict__ ahi,
                              const unsigned short* __restrict__ alo,
                              float* __restrict__ partial) {
    int wave = threadIdx.x >> 5;
    int lane = threadIdx.x & 31;
    int gw   = blockIdx.x * (blockDim.x >> 5) + wave;  // 0..511
    int ks   = gw >> 7;                                 // k-split slice
    int tile = gw & 127;
    int tm   = tile >> 4;
    int tn   = tile & 15;

    const v16bf* aH = ((const v16bf*)ahi) + (size_t)tm * KC_TOT * 32 + lane;
    const v16bf* aL = ((const v16bf*)alo) + (size_t)tm * KC_TOT * 32 + lane;
    const v16bf* wH = ((const v16bf*)whi) + (size_t)tn * KC_TOT * 32 + lane;
    const v16bf* wL = ((const v16bf*)wlo) + (size_t)tn * KC_TOT * 32 + lane;

    v8f c = {};
    int kc0 = ks * KC_PER;
#pragma unroll 2
    for (int kc = kc0; kc < kc0 + KC_PER; ++kc) {
        v16bf ah = aH[(size_t)kc * 32];
        v16bf al = aL[(size_t)kc * 32];
        v16bf wh = wH[(size_t)kc * 32];
        v16bf wl = wL[(size_t)kc * 32];
        c = __builtin_amdgcn_wmma_f32_16x16x32_bf16(false, ah, false, wh, (short)0, c, false, false);
        c = __builtin_amdgcn_wmma_f32_16x16x32_bf16(false, ah, false, wl, (short)0, c, false, false);
        c = __builtin_amdgcn_wmma_f32_16x16x32_bf16(false, al, false, wh, (short)0, c, false, false);
    }

    // C/D layout: VGPR r -> M = r + (lane>=16 ? 8 : 0), N = lane%16
    float* pp  = partial + ((size_t)(ks * NTILES + tile)) * 256;
    int rbase  = (lane >> 4) << 3;
    int col    = lane & 15;
#pragma unroll
    for (int r = 0; r < 8; ++r)
        pp[(rbase + r) * 16 + col] = c[r];
}

// ---------------------------------------------------------------------------
// Deterministic fixed-order split-K reduction -> out[128][256] f32
// ---------------------------------------------------------------------------
__global__ void kan_reduce(const float* __restrict__ partial,
                           float* __restrict__ out) {
    int idx = blockIdx.x * blockDim.x + threadIdx.x;   // 32768
    int b = idx >> 8;
    int n = idx & 255;
    int tile = (b >> 4) * 16 + (n >> 4);
    int off  = tile * 256 + (b & 15) * 16 + (n & 15);
    float s = 0.0f;
#pragma unroll
    for (int ks = 0; ks < KSPLIT; ++ks)
        s += partial[(size_t)ks * NTILES * 256 + off];
    out[idx] = s;
}

extern "C" void kernel_launch(void* const* d_in, const int* in_sizes, int n_in,
                              void* d_out, int out_size, void* d_ws, size_t ws_size,
                              hipStream_t stream) {
    const float* x    = (const float*)d_in[0];
    const float* grid = (const float*)d_in[1];
    const float* coef = (const float*)d_in[2];
    const float* rw   = (const float*)d_in[3];
    const float* uw   = (const float*)d_in[4];
    float* out = (float*)d_out;

    char* ws = (char*)d_ws;
    unsigned short* whi = (unsigned short*)(ws + OFF_WHI);
    unsigned short* wlo = (unsigned short*)(ws + OFF_WLO);
    unsigned short* ahi = (unsigned short*)(ws + OFF_AHI);
    unsigned short* alo = (unsigned short*)(ws + OFF_ALO);
    float*          prt = (float*)(ws + OFF_PART);

    kan_prep_w<<<(OUT_D * IN_D) / 256, 256, 0, stream>>>(coef, rw, uw, whi, wlo);
    kan_prep_a<<<(B_ROWS * IN_D) / 256, 256, 0, stream>>>(x, grid, ahi, alo);
    kan_wmma_gemm<<<(KSPLIT * NTILES) / 8, 256, 0, stream>>>(whi, wlo, ahi, alo, prt);
    kan_reduce<<<(B_ROWS * OUT_D) / 256, 256, 0, stream>>>(prt, out);
}
